// HybridNodeEdgeGraphTransformer_91250875171741
// MI455X (gfx1250) — matrix operand order repaired and checked by reading, not compiled, not run on hardware
//
#include <hip/hip_runtime.h>
#include <math.h>

// ---------------------------------------------------------------------------
// Types for CDNA5 WMMA (wave32, bf16 16x16x32, f32 accumulate)
// ---------------------------------------------------------------------------
typedef __attribute__((ext_vector_type(16))) __bf16 bf16x16;
typedef __attribute__((ext_vector_type(8)))  __bf16 bf16x8;
typedef __attribute__((ext_vector_type(8)))  float  f32x8;

static constexpr int B_   = 2;
static constexpr int N_   = 384;
static constexpr int D_   = 128;
static constexpr int NH_  = 8;
static constexpr int L_   = 2;
static constexpr int FN_  = 6;
static constexpr int FE_  = 4;
static constexpr int HID_ = 512;
static constexpr int NN_  = N_ * N_;
static constexpr int BN_  = B_ * N_;
static constexpr int BNN_ = B_ * NN_;

enum {
  EPI_STORE = 0,     // y = acc + bias
  EPI_BIAS_CM = 1,   // y = (acc + bias) * candidate_mask[row]   (attn bias)
  EPI_MPGATE = 2,    // fused sigmoid*tanh gate + masked j-reduction -> atomic S
  EPI_AGG = 3,       // y = (acc + mb[col]*sumcm[row]) * invdeg[row]
  EPI_GELU = 4,      // y = gelu(acc + bias)
  EPI_ADD = 5,       // y = node_x + acc + bias
  EPI_ADD_NVM = 6,   // y = node_x + (acc+bias)*nvm[row]
  EPI_RESID_NVM = 7  // y = (node_x + acc + bias)*nvm[row]
};

// ---------------------------------------------------------------------------
// Block-wide LayerNorm helper (blockDim.x == 128 == D)
// ---------------------------------------------------------------------------
static __device__ __forceinline__ float block_ln(float v, int c, float* sh) {
  sh[c] = v; __syncthreads();
  for (int s = 64; s > 0; s >>= 1) { if (c < s) sh[c] += sh[c + s]; __syncthreads(); }
  float m = sh[0] * (1.0f / 128.0f); __syncthreads();
  float d = v - m;
  sh[c] = d * d; __syncthreads();
  for (int s = 64; s > 0; s >>= 1) { if (c < s) sh[c] += sh[c + s]; __syncthreads(); }
  float var = sh[0] * (1.0f / 128.0f); __syncthreads();
  return d * rsqrtf(var + 1e-5f);
}

// ---------------------------------------------------------------------------
// Weight pack: f32 [K,Nc] row-major -> bf16 dense-B WMMA fragment order
// out[((nt*KT + kt)*32 + lane)*16 + e] = W[kt*32 + (e>>3)*16 + (lane>>4)*8 + (e&7), nt*16 + (lane&15)]
// ---------------------------------------------------------------------------
__global__ void pack_b_kernel(const float* __restrict__ W, __bf16* __restrict__ out,
                              int K, int Nc, long total) {
  long idx = (long)blockIdx.x * 256 + threadIdx.x;
  if (idx >= total) return;
  int e = (int)(idx & 15);
  int lane = (int)((idx >> 4) & 31);
  long rest = idx >> 9;
  int KT = K >> 5;
  int kt = (int)(rest % KT);
  int nt = (int)(rest / KT);
  int k = kt * 32 + (e >> 3) * 16 + (lane >> 4) * 8 + (e & 7);
  int col = nt * 16 + (lane & 15);
  float v = (col < Nc) ? W[(size_t)k * Nc + col] : 0.0f;
  out[idx] = (__bf16)v;
}

__global__ void f2b_kernel(const float* __restrict__ x, __bf16* __restrict__ y, long n) {
  long i = (long)blockIdx.x * 256 + threadIdx.x;
  if (i < n) y[i] = (__bf16)x[i];
}

__global__ void zero_kernel(float* __restrict__ x, long n) {
  long i = (long)blockIdx.x * 256 + threadIdx.x;
  if (i < n) x[i] = 0.0f;
}

// ---------------------------------------------------------------------------
// Node input encoder + LN  -> node_x (f32, lives in d_out)
// ---------------------------------------------------------------------------
__global__ __launch_bounds__(128) void node_enc_kernel(
    const float* __restrict__ H, const unsigned char* __restrict__ vis,
    const int* __restrict__ bus, const float* __restrict__ vw,
    const float* __restrict__ vb, const float* __restrict__ fme,
    const float* __restrict__ bte, const float* __restrict__ pe,
    const float* __restrict__ gam, const float* __restrict__ bet,
    float* __restrict__ node_x) {
  __shared__ float sh[128];
  const int row = blockIdx.x;       // b*N + i
  const int i = row % N_;
  const int c = threadIdx.x;
  float x = vb[c];
#pragma unroll
  for (int f = 0; f < FN_; ++f) {
    float vf = (float)vis[(size_t)row * FN_ + f];
    float hf = H[(size_t)row * FN_ + f];
    x += hf * vf * vw[f * D_ + c] + (1.0f - vf) * fme[f * D_ + c];
  }
  x += bte[bus[i] * D_ + c] + pe[i * D_ + c];
  float nrm = block_ln(x, c, sh);
  node_x[(size_t)row * D_ + c] = nrm * gam[c] + bet[c];
}

// ---------------------------------------------------------------------------
// Edge input encoder + LN + candidate mask -> ex (bf16)
// ---------------------------------------------------------------------------
__global__ __launch_bounds__(128) void edge_enc_kernel(
    const float* __restrict__ ef, const int* __restrict__ btype,
    const int* __restrict__ bstat, const float* __restrict__ epe,
    const unsigned char* __restrict__ cand, const float* __restrict__ vw,
    const float* __restrict__ vb, const float* __restrict__ bte,
    const float* __restrict__ bse, const float* __restrict__ gam,
    const float* __restrict__ bet, __bf16* __restrict__ exb) {
  __shared__ float sh[128];
  const long row = blockIdx.x;      // b*N*N + i*N + j
  const int ij = (int)(row % NN_);
  const int c = threadIdx.x;
  float x = vb[c];
#pragma unroll
  for (int f = 0; f < FE_; ++f)
    x += ef[row * FE_ + f] * vw[f * D_ + c];
  x += bte[btype[ij] * D_ + c] + bse[bstat[row] * D_ + c] + epe[row * D_ + c];
  float nrm = block_ln(x, c, sh);
  float y = (nrm * gam[c] + bet[c]) * (float)cand[row];
  exb[row * D_ + c] = (__bf16)y;
}

// ---------------------------------------------------------------------------
// LayerNorm over node_x -> bf16
// ---------------------------------------------------------------------------
__global__ __launch_bounds__(128) void ln_kernel(
    const float* __restrict__ x, const float* __restrict__ gam,
    const float* __restrict__ bet, __bf16* __restrict__ out) {
  __shared__ float sh[128];
  const int row = blockIdx.x;
  const int c = threadIdx.x;
  float v = x[(size_t)row * D_ + c];
  float nrm = block_ln(v, c, sh);
  out[(size_t)row * D_ + c] = (__bf16)(nrm * gam[c] + bet[c]);
}

// ---------------------------------------------------------------------------
// Closed-mask degree: sumcm[b,i] and 1/clip(sumcm,1)
// ---------------------------------------------------------------------------
__global__ __launch_bounds__(128) void degsum_kernel(
    const unsigned char* __restrict__ closed, float* __restrict__ sumcm,
    float* __restrict__ invdeg) {
  __shared__ float sh[128];
  const int r = blockIdx.x;
  const int t = threadIdx.x;
  float s = 0.0f;
  for (int j = t; j < N_; j += 128) s += (float)closed[(size_t)r * N_ + j];
  sh[t] = s; __syncthreads();
  for (int st = 64; st > 0; st >>= 1) { if (t < st) sh[t] += sh[t + st]; __syncthreads(); }
  if (t == 0) {
    float sm = sh[0];
    sumcm[r] = sm;
    invdeg[r] = 1.0f / fmaxf(sm, 1.0f);
  }
}

// ---------------------------------------------------------------------------
// WMMA GEMM:  C[M,Nc] = A[M,K](bf16,row-major) @ Bpacked[K,Nc](bf16) + epilogue
// Templated on KT (= K/32) and NTN (column tiles per wave) so the inner loop
// is branch-free straight-line loads + WMMAs with immediate address offsets.
// One wave computes a 16 x (16*NTN) strip; A fragment reused across NTN tiles.
// ---------------------------------------------------------------------------
template <int KT, int NTN>
__global__ __launch_bounds__(256) void gemm_bf16_wmma(
    const __bf16* __restrict__ A, const __bf16* __restrict__ Bp,
    const float* __restrict__ bias, float* __restrict__ outF,
    __bf16* __restrict__ outB, int M, int Nc, int mode,
    const float* __restrict__ auxRowA,   // per-row: sumcm
    const float* __restrict__ auxRowB,   // per-row: invdeg
    const float* __restrict__ auxT,      // tensor: node_j / node_x / per-col mb
    const __bf16* __restrict__ auxE,     // tensor: edge_term (bf16)
    const unsigned char* __restrict__ mask8, // per-row mask (cand/closed/nvm)
    float* __restrict__ Sacc,            // atomic reduction target
    int ldo, int ncolOut) {
  constexpr int K = KT * 32;
  const int Nt = Nc >> 4;
  const int G = (Nt + NTN - 1) / NTN;
  const int wave = blockIdx.x * 8 + (threadIdx.x >> 5);
  const int mt = wave / G;
  if (mt >= (M >> 4)) return;                 // wave-uniform exit
  const int g = wave - mt * G;
  const int nt0 = g * NTN;                    // all groups are full by dispatch
  const int lane = threadIdx.x & 31;
  const int half = lane >> 4;
  const int l15 = lane & 15;

  f32x8 acc[NTN];
#pragma unroll
  for (int t = 0; t < NTN; ++t)
#pragma unroll
    for (int e = 0; e < 8; ++e) acc[t][e] = 0.0f;

  const __bf16* arow = A + (size_t)(mt * 16 + l15) * K + half * 8;
  const __bf16* bbase = Bp + ((size_t)nt0 * KT * 32 + lane) * 16;
  for (int ko = 0; ko < KT / 4; ++ko) {
#pragma unroll
    for (int kk = 0; kk < 4; ++kk) {
      bf16x8 lo = *(const bf16x8*)(arow + kk * 32);
      bf16x8 hi = *(const bf16x8*)(arow + kk * 32 + 16);
      bf16x16 af;
#pragma unroll
      for (int e = 0; e < 8; ++e) { af[e] = lo[e]; af[e + 8] = hi[e]; }
#pragma unroll
      for (int t = 0; t < NTN; ++t) {
        bf16x16 bfrag = *(const bf16x16*)(bbase + kk * 512 + t * (KT * 512));
        acc[t] = __builtin_amdgcn_wmma_f32_16x16x32_bf16(
            false, af, false, bfrag, (short)0, acc[t], false, false);
      }
    }
    arow += 128;       // 4 k-steps * 32
    bbase += 4 * 512;  // 4 k-steps * (32 lanes * 16 elems)
  }

  // ---- epilogue ----
  const int rowBase = mt * 16 + half * 8;
#pragma unroll
  for (int t = 0; t < NTN; ++t) {
    const int col = (nt0 + t) * 16 + l15;
    const float bv = bias ? bias[col] : 0.0f;
    if (mode == EPI_MPGATE) {
      // 16 rows of this tile share (b,i); j varies (j innermost, 16 | N).
      const int bi = (mt * 16) / N_;
      const int b = (mt * 16) / NN_;
      float local = 0.0f;
#pragma unroll
      for (int r = 0; r < 8; ++r) {
        const int row = rowBase + r;
        const int j = row - (row / N_) * N_;
        float gate = 1.0f / (1.0f + __expf(-(acc[t][r] + bv)));
        float et = (float)auxE[(size_t)row * D_ + col];
        float nj = auxT[(size_t)(b * N_ + j) * D_ + col];
        local += tanhf(et + nj) * gate * (float)mask8[row];
      }
      local += __shfl_xor(local, 16);
      if (half == 0) atomicAdd(&Sacc[(size_t)bi * D_ + col], local);
      continue;
    }
#pragma unroll
    for (int r = 0; r < 8; ++r) {
      const int row = rowBase + r;
      float v = acc[t][r] + bv;
      float y;
      switch (mode) {
        default:             y = v; break;
        case EPI_BIAS_CM:    y = v * (float)mask8[row]; break;
        case EPI_AGG:        y = (v + auxT[col] * auxRowA[row]) * auxRowB[row]; break;
        case EPI_GELU:       y = 0.5f * v * (1.0f + erff(v * 0.70710678118654752f)); break;
        case EPI_ADD:        y = auxT[(size_t)row * ldo + col] + v; break;
        case EPI_ADD_NVM:    y = auxT[(size_t)row * ldo + col] + v * (float)mask8[row]; break;
        case EPI_RESID_NVM:  y = (auxT[(size_t)row * ldo + col] + v) * (float)mask8[row]; break;
      }
      if (col < ncolOut) {
        if (outF) outF[(size_t)row * ldo + col] = y;
        if (outB) outB[(size_t)row * ldo + col] = (__bf16)y;
      }
    }
  }
}

// ---------------------------------------------------------------------------
// Attention: one block per (b, h, i). hd = 16, N = 384 keys.
// ---------------------------------------------------------------------------
__global__ __launch_bounds__(384) void attn_kernel(
    const float* __restrict__ qkv, const float* __restrict__ bias,
    const unsigned char* __restrict__ nv, __bf16* __restrict__ out) {
  __shared__ float qs[16];
  __shared__ float sc[384];
  __shared__ float red[128];
  __shared__ float part[384];
  const int idx = blockIdx.x;
  const int i = idx % N_;
  const int h = (idx / N_) % NH_;
  const int b = idx / (N_ * NH_);
  const int t = threadIdx.x;
  const float* qrow = qkv + (size_t)(b * N_ + i) * (3 * D_);
  if (t < 16) qs[t] = qrow[h * 16 + t];
  __syncthreads();

  const int j = t;
  const bool valid = nv[b * N_ + i] && nv[b * N_ + j];
  float s = -1e30f;
  if (valid) {
    const float* krow = qkv + (size_t)(b * N_ + j) * (3 * D_) + D_ + h * 16;
    float d = 0.0f;
#pragma unroll
    for (int e = 0; e < 16; ++e) d += qs[e] * krow[e];
    s = d * 0.25f + bias[((size_t)(b * N_ + i) * N_ + j) * NH_ + h];
  }
  sc[t] = s; __syncthreads();
  if (t < 128) {
    float m = fmaxf(sc[t], fmaxf(sc[t + 128], sc[t + 256]));
    red[t] = m;
  }
  __syncthreads();
  for (int st = 64; st > 0; st >>= 1) { if (t < st) red[t] = fmaxf(red[t], red[t + st]); __syncthreads(); }
  const float M = red[0]; __syncthreads();

  float p = (s <= -1e29f || M <= -1e29f) ? 0.0f : __expf(s - M);
  sc[t] = p; __syncthreads();
  if (t < 128) red[t] = sc[t] + sc[t + 128] + sc[t + 256];
  __syncthreads();
  for (int st = 64; st > 0; st >>= 1) { if (t < st) red[t] += red[t + st]; __syncthreads(); }
  const float Z = red[0];
  const float invZ = (Z > 0.0f) ? (1.0f / Z) : 0.0f;  // nan_to_num semantics
  __syncthreads();

  const int dd = t & 15;
  const int gg = t >> 4;   // 0..23
  float a = 0.0f;
  for (int jj = gg; jj < N_; jj += 24)
    a += sc[jj] * qkv[(size_t)(b * N_ + jj) * (3 * D_) + 2 * D_ + h * 16 + dd];
  part[t] = a; __syncthreads();
  if (t < 16) {
    float o = 0.0f;
#pragma unroll
    for (int gq = 0; gq < 24; ++gq) o += part[gq * 16 + t];
    out[(size_t)(b * N_ + i) * D_ + h * 16 + t] = (__bf16)(o * invZ);
  }
}

// ---------------------------------------------------------------------------
// Host launcher
// ---------------------------------------------------------------------------
static inline void gemm(hipStream_t st, const __bf16* A, const __bf16* Bp,
                        const float* bias, float* outF, __bf16* outB,
                        int M, int K, int Nc, int mode,
                        const float* auxRowA = nullptr, const float* auxRowB = nullptr,
                        const float* auxT = nullptr, const __bf16* auxE = nullptr,
                        const unsigned char* mask8 = nullptr, float* Sacc = nullptr,
                        int ldo = -1, int ncolOut = -1) {
  if (ldo < 0) ldo = Nc;
  if (ncolOut < 0) ncolOut = Nc;
  const int Mt = M / 16, Nt = Nc / 16;
  const int ntn = (Nt < 8) ? Nt : 8;   // in this model: 8 or 1, groups always full
  const int G = (Nt + ntn - 1) / ntn;
  const long waves = (long)Mt * G;
  const int blocks = (int)((waves + 7) / 8);
  if (K == 512) {
    gemm_bf16_wmma<16, 8><<<blocks, 256, 0, st>>>(A, Bp, bias, outF, outB, M, Nc, mode,
                                                  auxRowA, auxRowB, auxT, auxE, mask8,
                                                  Sacc, ldo, ncolOut);
  } else if (ntn == 8) {
    gemm_bf16_wmma<4, 8><<<blocks, 256, 0, st>>>(A, Bp, bias, outF, outB, M, Nc, mode,
                                                 auxRowA, auxRowB, auxT, auxE, mask8,
                                                 Sacc, ldo, ncolOut);
  } else {
    gemm_bf16_wmma<4, 1><<<blocks, 256, 0, st>>>(A, Bp, bias, outF, outB, M, Nc, mode,
                                                 auxRowA, auxRowB, auxT, auxE, mask8,
                                                 Sacc, ldo, ncolOut);
  }
}

extern "C" void kernel_launch(void* const* d_in, const int* in_sizes, int n_in,
                              void* d_out, int out_size, void* d_ws, size_t ws_size,
                              hipStream_t stream) {
  (void)in_sizes; (void)n_in; (void)out_size; (void)ws_size;
  // ---- inputs ----
  const float* H    = (const float*)d_in[0];
  const float* ef   = (const float*)d_in[1];
  const float* epe  = (const float*)d_in[2];
  const int* bus    = (const int*)d_in[3];
  const int* btype  = (const int*)d_in[4];
  const int* bstat  = (const int*)d_in[5];
  const unsigned char* vis  = (const unsigned char*)d_in[6];
  const unsigned char* cand = (const unsigned char*)d_in[7];
  const unsigned char* clos = (const unsigned char*)d_in[8];
  const unsigned char* nv   = (const unsigned char*)d_in[9];
  const float* ne_vw = (const float*)d_in[10];
  const float* ne_vb = (const float*)d_in[11];
  const float* ne_fme = (const float*)d_in[12];
  const float* ne_bte = (const float*)d_in[13];
  const float* ne_pe = (const float*)d_in[14];
  const float* ne_ng = (const float*)d_in[15];
  const float* ne_nb = (const float*)d_in[16];
  const float* ee_vw = (const float*)d_in[17];
  const float* ee_vb = (const float*)d_in[18];
  const float* ee_bte = (const float*)d_in[19];
  const float* ee_bse = (const float*)d_in[20];
  const float* ee_bw = (const float*)d_in[21];
  const float* ee_bb = (const float*)d_in[22];
  const float* ee_ng = (const float*)d_in[23];
  const float* ee_nb = (const float*)d_in[24];
  const float* ln1_g = (const float*)d_in[25];
  const float* ln1_b = (const float*)d_in[26];
  const float* mp_nw = (const float*)d_in[27];
  const float* mp_nb = (const float*)d_in[28];
  const float* mp_ew = (const float*)d_in[29];
  const float* mp_eb = (const float*)d_in[30];
  const float* mp_mw = (const float*)d_in[31];
  const float* mp_mb = (const float*)d_in[32];
  const float* mp_gw = (const float*)d_in[33];
  const float* mp_gb = (const float*)d_in[34];
  const float* mp_o1w = (const float*)d_in[35];
  const float* mp_o1b = (const float*)d_in[36];
  const float* mp_o2w = (const float*)d_in[37];
  const float* mp_o2b = (const float*)d_in[38];
  const float* ln2_g = (const float*)d_in[39];
  const float* ln2_b = (const float*)d_in[40];
  const float* qkv_w = (const float*)d_in[41];
  const float* qkv_b = (const float*)d_in[42];
  const float* ao_w = (const float*)d_in[43];
  const float* ao_b = (const float*)d_in[44];
  const float* ln3_g = (const float*)d_in[45];
  const float* ln3_b = (const float*)d_in[46];
  const float* f1w = (const float*)d_in[47];
  const float* f1b = (const float*)d_in[48];
  const float* f2w = (const float*)d_in[49];
  const float* f2b = (const float*)d_in[50];

  float* node_x = (float*)d_out;   // [B,N,D] lives in the output buffer

  // ---- workspace carve-out ----
  size_t off = 0;
  auto alloc = [&](size_t bytes) -> void* {
    void* p = (char*)d_ws + off;
    off += (bytes + 255) & ~(size_t)255;
    return p;
  };
  __bf16* exb   = (__bf16*)alloc((size_t)BNN_ * D_ * 2);
  __bf16* etb   = (__bf16*)alloc((size_t)BNN_ * D_ * 2);
  float*  biasb = (float*)alloc((size_t)BNN_ * NH_ * 4);
  __bf16* h_bf  = (__bf16*)alloc((size_t)BN_ * D_ * 2);
  float*  nodej = (float*)alloc((size_t)BN_ * D_ * 4);
  float*  S     = (float*)alloc((size_t)BN_ * D_ * 4);
  __bf16* S_bf  = (__bf16*)alloc((size_t)BN_ * D_ * 2);
  __bf16* agg_bf = (__bf16*)alloc((size_t)BN_ * D_ * 2);
  __bf16* g1_bf = (__bf16*)alloc((size_t)BN_ * D_ * 2);
  float*  qkvb  = (float*)alloc((size_t)BN_ * 3 * D_ * 4);
  __bf16* at_bf = (__bf16*)alloc((size_t)BN_ * D_ * 2);
  __bf16* ff_bf = (__bf16*)alloc((size_t)BN_ * HID_ * 2);
  float*  sumcm = (float*)alloc((size_t)BN_ * 4);
  float*  invdeg = (float*)alloc((size_t)BN_ * 4);
  // packed weight buffers (reused across layers; stream serializes)
  __bf16* peebw = (__bf16*)alloc((size_t)2048 * 2);
  __bf16* pnw = (__bf16*)alloc((size_t)16384 * 2);
  __bf16* pew = (__bf16*)alloc((size_t)16384 * 2);
  __bf16* pgw = (__bf16*)alloc((size_t)16384 * 2);
  __bf16* pmw = (__bf16*)alloc((size_t)16384 * 2);
  __bf16* po1 = (__bf16*)alloc((size_t)16384 * 2);
  __bf16* po2 = (__bf16*)alloc((size_t)16384 * 2);
  __bf16* pqkv = (__bf16*)alloc((size_t)49152 * 2);
  __bf16* pao = (__bf16*)alloc((size_t)16384 * 2);
  __bf16* pf1 = (__bf16*)alloc((size_t)65536 * 2);
  __bf16* pf2 = (__bf16*)alloc((size_t)65536 * 2);

  auto pack = [&](const float* W, __bf16* out, int K, int Nc, int Ncp) {
    long total = (long)(Ncp / 16) * (K / 32) * 512;
    pack_b_kernel<<<(int)((total + 255) / 256), 256, 0, stream>>>(W, out, K, Nc, total);
  };

  // ---- encoders + shared precomputation ----
  node_enc_kernel<<<BN_, 128, 0, stream>>>(H, vis, bus, ne_vw, ne_vb, ne_fme,
                                           ne_bte, ne_pe, ne_ng, ne_nb, node_x);
  edge_enc_kernel<<<BNN_, 128, 0, stream>>>(ef, btype, bstat, epe, cand, ee_vw,
                                            ee_vb, ee_bte, ee_bse, ee_ng, ee_nb, exb);
  degsum_kernel<<<BN_, 128, 0, stream>>>(clos, sumcm, invdeg);

  pack(ee_bw, peebw, 128, 8, 16);
  // attn_bias[b,i,j,h] = (ex @ ee_bw + ee_bb) * cand
  gemm(stream, exb, peebw, ee_bb, biasb, nullptr, BNN_, 128, 16, EPI_BIAS_CM,
       nullptr, nullptr, nullptr, nullptr, cand, nullptr, /*ldo=*/NH_, /*ncolOut=*/NH_);

  // ---- layers ----
  for (int l = 0; l < L_; ++l) {
    pack(mp_nw + (size_t)l * D_ * D_, pnw, 128, 128, 128);
    pack(mp_ew + (size_t)l * D_ * D_, pew, 128, 128, 128);
    pack(mp_gw + (size_t)l * D_ * D_, pgw, 128, 128, 128);
    pack(mp_mw + (size_t)l * D_ * D_, pmw, 128, 128, 128);
    pack(mp_o1w + (size_t)l * D_ * D_, po1, 128, 128, 128);
    pack(mp_o2w + (size_t)l * D_ * D_, po2, 128, 128, 128);
    pack(qkv_w + (size_t)l * D_ * 3 * D_, pqkv, 128, 384, 384);
    pack(ao_w + (size_t)l * D_ * D_, pao, 128, 128, 128);
    pack(f1w + (size_t)l * D_ * HID_, pf1, 128, 512, 512);
    pack(f2w + (size_t)l * HID_ * D_, pf2, 512, 128, 128);

    // --- message passing ---
    ln_kernel<<<BN_, 128, 0, stream>>>(node_x, ln1_g + l * D_, ln1_b + l * D_, h_bf);
    gemm(stream, h_bf, pnw, mp_nb + l * D_, nodej, nullptr, BN_, 128, 128, EPI_STORE);
    gemm(stream, exb, pew, mp_eb + l * D_, nullptr, etb, BNN_, 128, 128, EPI_STORE);
    zero_kernel<<<(int)(((long)BN_ * D_ + 255) / 256), 256, 0, stream>>>(S, (long)BN_ * D_);
    // gate GEMM fused with tanh*sigmoid*closed-mask and sum_j reduction into S
    gemm(stream, etb, pgw, mp_gb + l * D_, nullptr, nullptr, BNN_, 128, 128, EPI_MPGATE,
         nullptr, nullptr, nodej, etb, clos, S);
    f2b_kernel<<<(int)(((long)BN_ * D_ + 255) / 256), 256, 0, stream>>>(S, S_bf, (long)BN_ * D_);
    // agg = (S @ mw + mb*sumcm) / deg
    gemm(stream, S_bf, pmw, nullptr, nullptr, agg_bf, BN_, 128, 128, EPI_AGG,
         sumcm, invdeg, mp_mb + l * D_);
    gemm(stream, agg_bf, po1, mp_o1b + l * D_, nullptr, g1_bf, BN_, 128, 128, EPI_GELU);
    gemm(stream, g1_bf, po2, mp_o2b + l * D_, node_x, nullptr, BN_, 128, 128, EPI_ADD_NVM,
         nullptr, nullptr, node_x, nullptr, nv);

    // --- global attention ---
    ln_kernel<<<BN_, 128, 0, stream>>>(node_x, ln2_g + l * D_, ln2_b + l * D_, h_bf);
    gemm(stream, h_bf, pqkv, qkv_b + l * 3 * D_, qkvb, nullptr, BN_, 128, 384, EPI_STORE);
    attn_kernel<<<B_ * NH_ * N_, 384, 0, stream>>>(qkvb, biasb, nv, at_bf);
    gemm(stream, at_bf, pao, ao_b + l * D_, node_x, nullptr, BN_, 128, 128, EPI_ADD,
         nullptr, nullptr, node_x);

    // --- feed forward ---
    ln_kernel<<<BN_, 128, 0, stream>>>(node_x, ln3_g + l * D_, ln3_b + l * D_, h_bf);
    gemm(stream, h_bf, pf1, f1b + l * HID_, nullptr, ff_bf, BN_, 128, 512, EPI_GELU);
    gemm(stream, ff_bf, pf2, f2b + l * D_, node_x, nullptr, BN_, 512, 128, EPI_RESID_NVM,
         nullptr, nullptr, node_x, nullptr, nv);
  }
}
